// SecondOrderDeformableAlignment_40973988004128
// MI455X (gfx1250) — matrix operand order repaired and checked by reading, not compiled, not run on hardware
//
#include <hip/hip_runtime.h>

// ---------------------------------------------------------------------------
// SecondOrderDeformableAlignment for MI455X (gfx1250, wave32, WMMA)
//   - all GEMMs via v_wmma_f32_16x16x32_f16 (f16 in, f32 accumulate)
//   - f16 intermediate activations (halves HBM traffic)
//   - weights pre-packed once into LDS-tile layout; staged with
//     global_load_async_to_lds_b128, DOUBLE-BUFFERED on ASYNCcnt so the
//     copy overlaps sampling + WMMA (threshold s_wait_asynccnt)
//   - halo-staged conv: 9 WMMAs per Cin-chunk, software-pipelined frag loads
//   - fast rcp (v_rcp_f32) for tanh/sigmoid reciprocals
// ---------------------------------------------------------------------------

#define USE_ASYNC_WMEM 1

typedef __attribute__((ext_vector_type(16))) _Float16 v16h;
typedef __attribute__((ext_vector_type(8)))  float    v8f;

#define HH 128
#define WW 128
#define HWSZ (HH*WW)          // 16384
#define BB 4

// ---- WMMA fragment loaders (ISA 7.12.2 layouts, wave32) --------------------
__device__ __forceinline__ v16h load_a_frag(const _Float16* base, int row, int hi) {
    union { v16h v; unsigned u[8]; } f;
    const unsigned* p = (const unsigned*)(base + row * 32);
    #pragma unroll
    for (int vv = 0; vv < 8; ++vv) {
        int k = ((vv & 4) << 2) + ((vv & 3) << 1) + (hi << 3);   // even
        f.u[vv] = p[k >> 1];
    }
    return f.v;
}
__device__ __forceinline__ v16h load_b_frag(const _Float16* base, int row, int hi) {
    union { v16h v; unsigned u[8]; } f;
    const unsigned* p = (const unsigned*)(base + row * 32);
    #pragma unroll
    for (int vv = 0; vv < 8; ++vv)
        f.u[vv] = p[((vv & 4) << 1) + (vv & 3) + (hi << 2)];
    return f.v;
}

// Issue `beats` 16B async copies global->LDS (no wait): ASYNCcnt path.
__device__ __forceinline__ void stage_tile_issue(const _Float16* __restrict__ gsrc,
                                                 _Float16* lds, int tid, int beats) {
#if USE_ASYNC_WMEM
    const unsigned ldsB = (unsigned)(unsigned long long)(const void*)lds;
    const unsigned long long ga = (unsigned long long)gsrc;
    for (int j = tid; j < beats; j += 256) {
        const unsigned off = (unsigned)j << 4;
        asm volatile("global_load_async_to_lds_b128 %0, %1, off"
                     :: "v"(ldsB + off), "v"(ga + off) : "memory");
    }
#else
    const uint4* s = (const uint4*)gsrc;
    uint4* d = (uint4*)lds;
    for (int j = tid; j < beats; j += 256) d[j] = s[j];
#endif
}
template <int N> __device__ __forceinline__ void wait_async() {
#if USE_ASYNC_WMEM
    asm volatile("s_wait_asynccnt %0" :: "n"(N) : "memory");
#endif
}

// ---------------------------------------------------------------------------
// Weight pre-pack (run once, tiny): conv tile = [tap][n][ci] f16,
// 4608 valid halves zero-padded to 6144 (12288 B) per (nTile, cc).
// ---------------------------------------------------------------------------
__global__ __launch_bounds__(256)
void pack_conv_w(const float* __restrict__ wgt, _Float16* __restrict__ outp,
                 int Cin, int Cout)
{
    const int ccN  = (Cin + 31) >> 5;
    const int tile = blockIdx.x;
    const int nt   = tile / ccN;
    const int cc   = tile - nt * ccN;
    for (int e = threadIdx.x; e < 6144; e += 256) {
        _Float16 v = (_Float16)0.0f;
        if (e < 4608) {
            const int tap = e >> 9;
            const int rem = e & 511;
            const int n   = rem >> 5;
            const int ci  = rem & 31;
            const int ciG = (cc << 5) + ci;
            if (ciG < Cin)
                v = (_Float16)wgt[((long)(nt * 16 + n) * Cin + ciG) * 9 + tap];
        }
        outp[(size_t)tile * 6144 + e] = v;
    }
}

// deform tile = [n][kk] f16, 2048 halves (4096 B) per chunk
__global__ __launch_bounds__(256)
void pack_deform_w(const float* __restrict__ dw, _Float16* __restrict__ outp)
{
    const int chunk = blockIdx.x;              // 36
    for (int e = threadIdx.x; e < 2048; e += 256) {
        const int n  = e >> 5;
        const int kk = e & 31;
        const int kp = (chunk << 5) + kk;
        const int c  = kp & 7;
        const int gk = kp >> 3;
        const int g  = gk / 9;
        const int k  = gk - g * 9;
        const int ci = (g << 3) + c;
        outp[(size_t)chunk * 2048 + e] = (_Float16)dw[((long)n * 128 + ci) * 9 + k];
    }
}

// ---------------------------------------------------------------------------
// Generic 3x3 SAME conv, implicit GEMM. TIN = float (layer 1) or _Float16.
// One block = one image row x 16 cout; f16 output.
// ---------------------------------------------------------------------------
template <typename TIN>
__global__ __launch_bounds__(256)
void conv3x3_wmma(const TIN* __restrict__ in, const _Float16* __restrict__ wpk,
                  const float* __restrict__ bias, _Float16* __restrict__ out,
                  int Cin, int Cout, int act)
{
    __shared__ __align__(16) unsigned char smem[24960 + 2 * 12288];
    _Float16* As3 = (_Float16*)smem;                    // 3*130*32 halves
    float*    Cs  = (float*)smem;                       // 16*128 f32 (reuse)

    const int tid  = threadIdx.x;
    const int lane = tid & 31;
    const int wave = tid >> 5;
    const int hi   = lane >> 4;
    const int blockM = blockIdx.x;                      // one image row
    const int nBase  = blockIdx.y << 4;
    const int b = blockM >> 7;
    const int h = blockM & 127;

    if (tid < 192) {   // zero halo columns p=-1, p=128 once
        const int dh   = tid >> 6;
        const int side = (tid >> 5) & 1;
        const int ci   = tid & 31;
        As3[(dh * 130 + (side ? 129 : 0)) * 32 + ci] = (_Float16)0.0f;
    }

    v8f acc = {};
    const int ccN = (Cin + 31) >> 5;
    const int rowA = (wave << 4) + (lane & 15);
    const _Float16* wbase = wpk + (size_t)blockIdx.y * ccN * 6144;

    stage_tile_issue(wbase, (_Float16*)(smem + 24960), tid, 768);   // tile 0

    for (int cc = 0; cc < ccN; ++cc) {
        _Float16* Bt9 = (_Float16*)(smem + 24960 + ((cc & 1) ? 12288 : 0));
        const bool more = (cc + 1 < ccN);
        if (more) {                                      // overlap copy w/ work
            _Float16* BtNext =
                (_Float16*)(smem + 24960 + (((cc + 1) & 1) ? 12288 : 0));
            stage_tile_issue(wbase + (size_t)(cc + 1) * 6144, BtNext, tid, 768);
        }

        // ---- stage A halo: 3 rows x 32 ci x 128 px ----
        if constexpr (sizeof(TIN) == 4) {
            const int pl = (tid & 31) << 2;             // 32 thr x 4 px (float4)
            const int cg = tid >> 5;
            #pragma unroll
            for (int r = 0; r < 12; ++r) {
                const int pair = cg + (r << 3);         // 0..95
                const int dh_i = pair >> 5;
                const int ci   = pair & 31;
                const int sy   = h + dh_i - 1;
                const int ciG  = (cc << 5) + ci;
                const float s  = ((sy >= 0) & (sy < HH) & (ciG < Cin)) ? 1.0f : 0.0f;
                const int syc  = min(max(sy, 0), HH - 1);
                const int cic  = min(ciG, Cin - 1);
                const float4 d = *(const float4*)((const float*)in
                        + ((long)b * Cin + cic) * HWSZ + syc * WW + pl);
                _Float16* dst = As3 + (dh_i * 130 + pl + 1) * 32 + ci;
                dst[0]  = (_Float16)(d.x * s);
                dst[32] = (_Float16)(d.y * s);
                dst[64] = (_Float16)(d.z * s);
                dst[96] = (_Float16)(d.w * s);
            }
        } else {
            const int pg8 = (tid & 15) << 3;            // 16 thr x 8 px (16B)
            const int grp = tid >> 4;
            #pragma unroll
            for (int r = 0; r < 6; ++r) {
                const int pair = grp + (r << 4);        // 0..95
                const int dh_i = pair >> 5;
                const int ci   = pair & 31;
                const int sy   = h + dh_i - 1;
                const int ciG  = (cc << 5) + ci;
                const float s  = ((sy >= 0) & (sy < HH) & (ciG < Cin)) ? 1.0f : 0.0f;
                const int syc  = min(max(sy, 0), HH - 1);
                const int cic  = min(ciG, Cin - 1);
                union { uint4 u; _Float16 hh[8]; } d;
                d.u = *(const uint4*)((const _Float16*)in
                        + ((long)b * Cin + cic) * HWSZ + syc * WW + pg8);
                _Float16* dst = As3 + (dh_i * 130 + pg8 + 1) * 32 + ci;
                #pragma unroll
                for (int q = 0; q < 8; ++q)
                    dst[q * 32] = (_Float16)((float)d.hh[q] * s);
            }
        }
        // older tile done once in-flight <= newer tile's 3 beats
        if (more) wait_async<3>(); else wait_async<0>();
        __syncthreads();

        // ---- 9 taps, software-pipelined fragment loads ----
        v16h a  = load_a_frag(As3, rowA, hi);           // tap0: dh=0,dw=-1
        v16h bf = load_b_frag(Bt9, lane & 15, hi);
        #pragma unroll
        for (int tap = 0; tap < 9; ++tap) {
            v16h a2 = a, b2 = bf;
            if (tap < 8) {
                const int t2  = tap + 1;
                const int dh2 = t2 / 3;
                const int dw2 = t2 % 3 - 1;
                a2 = load_a_frag(As3 + (dh2 * 130 + dw2 + 1) * 32, rowA, hi);
                b2 = load_b_frag(Bt9 + (t2 << 9), lane & 15, hi);
            }
            acc = __builtin_amdgcn_wmma_f32_16x16x32_f16(
                    false, a, false, bf, (short)0, acc, false, false);
            a = a2; bf = b2;
        }
        __syncthreads();
    }

    // ---- epilogue: LDS transpose -> packed coalesced f16 row writes ----
    #pragma unroll
    for (int r = 0; r < 8; ++r)
        Cs[(lane & 15) * 128 + (wave << 4) + r + (hi << 3)] = acc[r];
    __syncthreads();

    #pragma unroll
    for (int j = 0; j < 4; ++j) {
        const int lin = (j << 8) + tid;                 // 0..1023 (pairs)
        const int n   = lin >> 6;
        const int pp  = (lin & 63) << 1;
        const float bv = bias[nBase + n];
        float v0 = Cs[n * 128 + pp]     + bv;
        float v1 = Cs[n * 128 + pp + 1] + bv;
        if (act) {
            v0 = v0 > 0.0f ? v0 : 0.1f * v0;
            v1 = v1 > 0.0f ? v1 : 0.1f * v1;
        }
        union { _Float16 hh[2]; unsigned u; } o;
        o.hh[0] = (_Float16)v0; o.hh[1] = (_Float16)v1;
        *(unsigned*)(out + ((long)b * Cout + nBase + n) * HWSZ + (h << 7) + pp) = o.u;
    }
}

// ---------------------------------------------------------------------------
// Modulated deformable conv, fully fused (y4 f16, x f32, out f32).
// ---------------------------------------------------------------------------
__global__ __launch_bounds__(256)
void deform_wmma(const float* __restrict__ x,    const _Float16* __restrict__ y4,
                 const float* __restrict__ flow, const _Float16* __restrict__ wpk,
                 const float* __restrict__ db,   float* __restrict__ out)
{
    __shared__ __align__(16) unsigned char smem[32768];
    _Float16* As = (_Float16*)smem;                     // 128*32 halves = 8 KB
    float*    Cs = (float*)smem;                        // 64*128 f32 (reuse)

    const int tid  = threadIdx.x;
    const int lane = tid & 31;
    const int wave = tid >> 5;
    const int hi   = lane >> 4;
    const int waveN = wave & 3;
    const int waveM = wave >> 2;
    const int blockM = blockIdx.x;
    const int b = blockM >> 7;
    const int h = blockM & 127;

    v8f acc[4] = {};

    stage_tile_issue(wpk, (_Float16*)(smem + 8192), tid, 256);  // chunk 0

    for (int chunk = 0; chunk < 36; ++chunk) {
        _Float16* Bs = (_Float16*)(smem + 8192 + ((chunk & 1) ? 4096 : 0));
        const bool more = (chunk + 1 < 36);
        if (more) {
            _Float16* BsNext =
                (_Float16*)(smem + 8192 + (((chunk + 1) & 1) ? 4096 : 0));
            stage_tile_issue(wpk + (size_t)(chunk + 1) * 2048, BsNext, tid, 256);
        }

        // ---- stage A via bilinear sampling: 512 (pixel,tap) items ----
        #pragma unroll
        for (int halfit = 0; halfit < 2; ++halfit) {
            const int it = tid + (halfit << 8);
            const int p  = it >> 2;                     // pixel (= w)
            const int i  = it & 3;
            const int gk = (chunk << 2) + i;            // 0..143
            const int g  = gk / 9;
            const int k  = gk - g * 9;
            const int dy = k / 3 - 1, dx = k % 3 - 1;
            const int hw = (h << 7) + p;

            const _Float16* yp = y4 + (long)b * 432 * HWSZ + hw;
            const float oyr = (float)yp[(long)(gk * 2)     * HWSZ];
            const float oxr = (float)yp[(long)(gk * 2 + 1) * HWSZ];
            const float mr  = (float)yp[(long)(288 + gk)   * HWSZ];
            if (gk + 4 < 144) {   // prefetch next chunk's channels
                __builtin_prefetch(&yp[(long)((gk + 4) * 2) * HWSZ], 0, 0);
                __builtin_prefetch(&yp[(long)(288 + gk + 4) * HWSZ], 0, 0);
            }
            const float fy  = flow[((long)b * 2 + 1) * HWSZ + hw]; // flow[:, ::-1]
            const float fx  = flow[((long)b * 2 + 0) * HWSZ + hw];

            // 10*tanh(v) = 10 - 20*rcp(exp(2v)+1); sigmoid = rcp(1+exp(-v))
            const float oy  = 10.0f
                - 20.0f * __builtin_amdgcn_rcpf(__expf(2.0f * oyr) + 1.0f) + fy;
            const float ox  = 10.0f
                - 20.0f * __builtin_amdgcn_rcpf(__expf(2.0f * oxr) + 1.0f) + fx;
            const float msk = __builtin_amdgcn_rcpf(1.0f + __expf(-mr));

            const float py = oy + (float)(h + dy);
            const float px = ox + (float)(p + dx);
            const float y0f = floorf(py), x0f = floorf(px);
            const float wy = py - y0f,   wx = px - x0f;
            const int y0 = (int)y0f, x0 = (int)x0f;

            const int yi0 = min(max(y0, 0), HH - 1);
            const int yi1 = min(max(y0 + 1, 0), HH - 1);
            const int xi0 = min(max(x0, 0), WW - 1);
            const int xi1 = min(max(x0 + 1, 0), WW - 1);
            const float vy0 = ((y0 >= 0) & (y0 < HH)) ? 1.0f : 0.0f;
            const float vy1 = ((y0 + 1 >= 0) & (y0 + 1 < HH)) ? 1.0f : 0.0f;
            const float vx0 = ((x0 >= 0) & (x0 < WW)) ? 1.0f : 0.0f;
            const float vx1 = ((x0 + 1 >= 0) & (x0 + 1 < WW)) ? 1.0f : 0.0f;
            const float w00 = vy0 * vx0 * (1.0f - wy) * (1.0f - wx) * msk;
            const float w01 = vy0 * vx1 * (1.0f - wy) * wx          * msk;
            const float w10 = vy1 * vx0 * wy * (1.0f - wx)          * msk;
            const float w11 = vy1 * vx1 * wy * wx                   * msk;
            const int i00 = yi0 * WW + xi0, i01 = yi0 * WW + xi1;
            const int i10 = yi1 * WW + xi0, i11 = yi1 * WW + xi1;

            const long xbase = ((long)b * 128 + (g << 3)) * HWSZ;
            union { _Float16 hh[8]; uint4 u; } pk;
            #pragma unroll
            for (int c = 0; c < 8; ++c) {
                const long cb = xbase + (long)c * HWSZ;
                float s = w00 * x[cb + i00] + w01 * x[cb + i01]
                        + w10 * x[cb + i10] + w11 * x[cb + i11];
                pk.hh[c] = (_Float16)s;
            }
            *(uint4*)(As + p * 32 + (i << 3)) = pk.u;   // ds_store_b128
        }
        if (more) wait_async<1>(); else wait_async<0>();
        __syncthreads();

        v16h bfrag = load_b_frag(Bs + (waveN << 4) * 32, lane & 15, hi);
        v16h af[4];
        #pragma unroll
        for (int ms = 0; ms < 4; ++ms)
            af[ms] = load_a_frag(As + ((waveM << 6) + (ms << 4)) * 32,
                                 lane & 15, hi);
        #pragma unroll
        for (int ms = 0; ms < 4; ++ms)
            acc[ms] = __builtin_amdgcn_wmma_f32_16x16x32_f16(
                        false, af[ms], false, bfrag, (short)0, acc[ms],
                        false, false);
        __syncthreads();
    }

    // ---- epilogue: LDS transpose -> coalesced f32 row writes ----
    #pragma unroll
    for (int ms = 0; ms < 4; ++ms) {
        const int n = (waveN << 4) + (lane & 15);
        #pragma unroll
        for (int r = 0; r < 8; ++r) {
            const int p = (waveM << 6) + (ms << 4) + r + (hi << 3);
            Cs[n * 128 + p] = acc[ms][r];
        }
    }
    __syncthreads();

    #pragma unroll
    for (int j = 0; j < 32; ++j) {
        const int lin = (j << 8) + tid;                 // 0..8191
        const int n   = lin >> 7;
        const int p   = lin & 127;
        out[((long)b * 64 + n) * HWSZ + (h << 7) + p] = Cs[lin] + db[n];
    }
}

// ---------------------------------------------------------------------------
extern "C" void kernel_launch(void* const* d_in, const int* in_sizes, int n_in,
                              void* d_out, int out_size, void* d_ws, size_t ws_size,
                              hipStream_t stream) {
    const float* x    = (const float*)d_in[0];
    const float* cond = (const float*)d_in[1];
    const float* flow = (const float*)d_in[2];
    const float* w1 = (const float*)d_in[3];
    const float* b1 = (const float*)d_in[4];
    const float* w2 = (const float*)d_in[5];
    const float* b2 = (const float*)d_in[6];
    const float* w3 = (const float*)d_in[7];
    const float* b3 = (const float*)d_in[8];
    const float* w4 = (const float*)d_in[9];
    const float* b4 = (const float*)d_in[10];
    const float* dw = (const float*)d_in[11];
    const float* db = (const float*)d_in[12];

    const size_t S64 = (size_t)BB * 64 * HWSZ;        // halves per 64-ch map
    const size_t SY4 = (size_t)BB * 432 * HWSZ;       // halves for y4
    _Float16* bufA = (_Float16*)d_ws;
    _Float16* bufB = bufA + S64;
    _Float16* y4   = bufB + S64;
    _Float16* pk1  = y4  + SY4;                       // 4*7  tiles * 6144
    _Float16* pk2  = pk1 + (size_t)28 * 6144;         // 4*2
    _Float16* pk3  = pk2 + (size_t)8  * 6144;         // 4*2
    _Float16* pk4  = pk3 + (size_t)8  * 6144;         // 27*2
    _Float16* pkd  = pk4 + (size_t)54 * 6144;         // 36 * 2048

    dim3 blk(256);
    pack_conv_w  <<<dim3(28), blk, 0, stream>>>(w1, pk1, 196, 64);
    pack_conv_w  <<<dim3(8),  blk, 0, stream>>>(w2, pk2,  64, 64);
    pack_conv_w  <<<dim3(8),  blk, 0, stream>>>(w3, pk3,  64, 64);
    pack_conv_w  <<<dim3(54), blk, 0, stream>>>(w4, pk4,  64, 432);
    pack_deform_w<<<dim3(36), blk, 0, stream>>>(dw, pkd);

    conv3x3_wmma<float>   <<<dim3(512, 4),  blk, 0, stream>>>(cond, pk1, b1, bufA, 196, 64, 1);
    conv3x3_wmma<_Float16><<<dim3(512, 4),  blk, 0, stream>>>(bufA, pk2, b2, bufB,  64, 64, 1);
    conv3x3_wmma<_Float16><<<dim3(512, 4),  blk, 0, stream>>>(bufB, pk3, b3, bufA,  64, 64, 1);
    conv3x3_wmma<_Float16><<<dim3(512, 27), blk, 0, stream>>>(bufA, pk4, b4, y4,    64, 432, 0);
    deform_wmma           <<<dim3(512),     blk, 0, stream>>>(x, y4, flow, pkd, db, (float*)d_out);
}
